// SimpleRNN_86569360818945
// MI455X (gfx1250) — matrix operand order repaired and checked by reading, not compiled, not run on hardware
//
#include <hip/hip_runtime.h>
#include <hip/hip_bf16.h>

// ---- shapes -------------------------------------------------------------
#define B_   64
#define T_   512
#define KIN  512
#define H_   1024
#define O_   512

typedef __bf16 v16bf __attribute__((ext_vector_type(16)));
typedef __bf16 v8bf  __attribute__((ext_vector_type(8)));
typedef float  v8f   __attribute__((ext_vector_type(8)));

static __device__ __forceinline__ v8f wmma_bf16(v16bf a, v16bf b, v8f c) {
  // D = A(16x32 bf16) * B(32x16 bf16) + C(16x16 f32)
  return __builtin_amdgcn_wmma_f32_16x16x32_bf16(false, a, false, b,
                                                 (short)0, c, false, false);
}

// ---- pack row-major f32 [K][N] into WMMA-B bf16 fragments ---------------
// packed[((ntile*KT + ktile)*32 + lane)*16 + i] = bf16(src[k*N + n])
//   k = ktile*32 + (lane>>4)*8 + (i<8 ? i : i+8) ;  n = ntile*16 + (lane&15)
__global__ void pack_b_bf16(const float* __restrict__ src,
                            __bf16* __restrict__ dst, int K, int N) {
  int idx = blockIdx.x * blockDim.x + threadIdx.x;
  if (idx >= K * N) return;
  int i    = idx & 15;
  int lane = (idx >> 4) & 31;
  int tile = idx >> 9;
  int KT   = K >> 5;
  int ktile = tile % KT;
  int ntile = tile / KT;
  int k = ktile * 32 + ((lane >> 4) << 3) + ((i < 8) ? i : i + 8);
  int n = ntile * 16 + (lane & 15);
  dst[idx] = (__bf16)src[k * N + n];
}

// ---- pack row-major f32 [M][K] into WMMA-A bf16 fragments ---------------
// packed[((mtile*KT + ktile)*32 + lane)*16 + i] = bf16(src[m*K + k])
//   m = mtile*16 + (lane&15) ;  k = ktile*32 + (lane>>4)*8 + (i<8 ? i : i+8)
__global__ void pack_a_bf16(const float* __restrict__ src,
                            __bf16* __restrict__ dst, int M, int K) {
  int idx = blockIdx.x * blockDim.x + threadIdx.x;
  if (idx >= M * K) return;
  int i    = idx & 15;
  int lane = (idx >> 4) & 31;
  int tile = idx >> 9;
  int KT   = K >> 5;
  int ktile = tile % KT;
  int mtile = tile / KT;
  int m = mtile * 16 + (lane & 15);
  int k = ktile * 32 + ((lane >> 4) << 3) + ((i < 8) ? i : i + 8);
  dst[idx] = (__bf16)src[(size_t)m * K + k];
}

// ---- kernel 1: XU[m][n] = X[m][k] @ U[k][n] + b[n]  (M=32768,K=512,N=1024)
// Pure-fragment GEMM: no LDS, no barriers, no conversions.
// 256 threads = 8 waves arranged 2(M) x 4(N); wave tile 64x32; WG tile 128x128.
__global__ __launch_bounds__(256) void xu_gemm(const __bf16* __restrict__ Xpack,
                                               const __bf16* __restrict__ Upack,
                                               const float* __restrict__ bias,
                                               float* __restrict__ XU) {
  const int tid  = threadIdx.x;
  const int lane = tid & 31;
  const int wave = tid >> 5;
  const int wm   = wave & 1;             // 2 in M
  const int wn   = wave >> 1;            // 4 in N
  const int l16  = lane & 15;
  const int kgrp = lane >> 4;
  const int mBase = blockIdx.x * 128 + wm * 64;
  const int nBase = blockIdx.y * 128 + wn * 32;
  const int KT = KIN / 32;               // 16 k-steps

  v8f acc[4][2] = {};

  for (int ks = 0; ks < KT; ++ks) {
    v16bf b0 = *(const v16bf*)(Upack + ((size_t)(((nBase >> 4) + 0) * KT + ks) * 32 + lane) * 16);
    v16bf b1 = *(const v16bf*)(Upack + ((size_t)(((nBase >> 4) + 1) * KT + ks) * 32 + lane) * 16);
#pragma unroll
    for (int mi = 0; mi < 4; ++mi) {
      v16bf a = *(const v16bf*)(Xpack + ((size_t)(((mBase >> 4) + mi) * KT + ks) * 32 + lane) * 16);
      acc[mi][0] = wmma_bf16(a, b0, acc[mi][0]);
      acc[mi][1] = wmma_bf16(a, b1, acc[mi][1]);
    }
  }

#pragma unroll
  for (int mi = 0; mi < 4; ++mi)
#pragma unroll
    for (int ni = 0; ni < 2; ++ni) {
      int n  = nBase + ni * 16 + l16;
      float bv = bias[n];
      int m0 = mBase + mi * 16 + kgrp * 8;
#pragma unroll
      for (int j = 0; j < 8; ++j)
        XU[(size_t)(m0 + j) * H_ + n] = acc[mi][ni][j] + bv;
    }
}

// ---- kernel 2: sequential scan ------------------------------------------
// 4 WGs (16 batch rows each) x 1024 threads (32 waves).
// Per step: z = h@W + xu_t (wave owns 32 of 1024 cols), tanh -> h (LDS bf16),
// y = h@V + c (wave owns 16 of 512 cols), per-row softmax (1 wave / row).
#define HP (H_ + 8)
#define OPAD (O_ + 8)

__global__ __launch_bounds__(1024) void rnn_scan(
    const float* __restrict__ h0,
    const __bf16* __restrict__ Wpack,
    const __bf16* __restrict__ Vpack,
    const float* __restrict__ XU,
    const float* __restrict__ cbias,
    float* __restrict__ Y) {
  __shared__ __bf16 hA[16][HP];          // h in A-fragment-friendly layout
  __shared__ float  yB[16][OPAD];        // pre-softmax logits

  const int tid  = threadIdx.x;
  const int lane = tid & 31;
  const int wave = tid >> 5;             // 0..31
  const int l16  = lane & 15;
  const int kgrp = lane >> 4;
  const int bBase = blockIdx.x * 16;
  const int KT = H_ / 32;                // 32 k-steps

  for (int idx = tid; idx < 16 * H_; idx += 1024) {
    int r = idx >> 10;
    int c = idx & (H_ - 1);
    hA[r][c] = (__bf16)h0[(bBase + r) * H_ + c];
  }
  __syncthreads();

  const float cv = cbias[wave * 16 + l16];

  for (int t = 0; t < T_; ++t) {
    // Issue the xu_t gather early; it is consumed only after the K-loop, so
    // ~64 WMMAs hide the load latency on the serial path.
    float xv0[8], xv1[8];
    {
      const float* xup = XU + ((size_t)(bBase + kgrp * 8) * T_ + t) * H_ + wave * 32 + l16;
#pragma unroll
      for (int j = 0; j < 8; ++j) {
        xv0[j] = xup[(size_t)j * T_ * H_];
        xv1[j] = xup[(size_t)j * T_ * H_ + 16];
      }
    }

    // -------- z = h @ W ---------------------------------------------------
    v8f acc0 = {}, acc1 = {};
#pragma unroll 4
    for (int ks = 0; ks < KT; ++ks) {
      const __bf16* rp = &hA[l16][ks * 32 + kgrp * 8];
      v8bf lo = *(const v8bf*)rp;
      v8bf hi = *(const v8bf*)(rp + 16);
      v16bf a;
#pragma unroll
      for (int e = 0; e < 8; ++e) { a[e] = lo[e]; a[e + 8] = hi[e]; }
      v16bf b0 = *(const v16bf*)(Wpack + ((size_t)((wave * 2 + 0) * KT + ks) * 32 + lane) * 16);
      v16bf b1 = *(const v16bf*)(Wpack + ((size_t)((wave * 2 + 1) * KT + ks) * 32 + lane) * 16);
      acc0 = wmma_bf16(a, b0, acc0);
      acc1 = wmma_bf16(a, b1, acc1);
    }
    __syncthreads();                       // everyone done reading old h
#pragma unroll
    for (int j = 0; j < 8; ++j) {          // h_new = tanh(z + xu) -> LDS
      hA[kgrp * 8 + j][wave * 32 + l16]      = (__bf16)tanhf(acc0[j] + xv0[j]);
      hA[kgrp * 8 + j][wave * 32 + 16 + l16] = (__bf16)tanhf(acc1[j] + xv1[j]);
    }
    __syncthreads();

    // -------- y = h_new @ V + c ------------------------------------------
    v8f yacc;
#pragma unroll
    for (int j = 0; j < 8; ++j) yacc[j] = cv;
#pragma unroll 4
    for (int ks = 0; ks < KT; ++ks) {
      const __bf16* rp = &hA[l16][ks * 32 + kgrp * 8];
      v8bf lo = *(const v8bf*)rp;
      v8bf hi = *(const v8bf*)(rp + 16);
      v16bf a;
#pragma unroll
      for (int e = 0; e < 8; ++e) { a[e] = lo[e]; a[e + 8] = hi[e]; }
      v16bf bv = *(const v16bf*)(Vpack + ((size_t)(wave * KT + ks) * 32 + lane) * 16);
      yacc = wmma_bf16(a, bv, yacc);
    }
#pragma unroll
    for (int j = 0; j < 8; ++j)
      yB[kgrp * 8 + j][wave * 16 + l16] = yacc[j];
    __syncthreads();

    // -------- softmax: wave r handles batch row r ------------------------
    if (wave < 16) {
      float ev[O_ / 32];
      float mx = -3.4e38f;
#pragma unroll
      for (int i = 0; i < O_ / 32; ++i) {
        float v = yB[wave][i * 32 + lane];
        ev[i] = v;
        mx = fmaxf(mx, v);
      }
#pragma unroll
      for (int off = 16; off > 0; off >>= 1)
        mx = fmaxf(mx, __shfl_xor(mx, off, 32));
      float sum = 0.f;
#pragma unroll
      for (int i = 0; i < O_ / 32; ++i) {
        ev[i] = __expf(ev[i] - mx);
        sum += ev[i];
      }
#pragma unroll
      for (int off = 16; off > 0; off >>= 1)
        sum += __shfl_xor(sum, off, 32);
      float inv = 1.0f / sum;
      float* yp = Y + ((size_t)(bBase + wave) * T_ + t) * O_;
#pragma unroll
      for (int i = 0; i < O_ / 32; ++i)
        yp[i * 32 + lane] = ev[i] * inv;
    }
    // yB is only rewritten after two barriers in iteration t+1, and hA only
    // after one -> no trailing barrier needed.
  }
}

// ---- host glue -----------------------------------------------------------
extern "C" void kernel_launch(void* const* d_in, const int* in_sizes, int n_in,
                              void* d_out, int out_size, void* d_ws, size_t ws_size,
                              hipStream_t stream) {
  const float* X  = (const float*)d_in[0];
  const float* h0 = (const float*)d_in[1];
  const float* W  = (const float*)d_in[2];
  const float* U  = (const float*)d_in[3];
  const float* bb = (const float*)d_in[4];
  const float* V  = (const float*)d_in[5];
  const float* cc = (const float*)d_in[6];
  float* Y = (float*)d_out;

  char* ws = (char*)d_ws;
  const size_t M1 = (size_t)B_ * T_;                 // 32768
  size_t off = 0;
  float*  XU    = (float*)(ws + off); off += M1 * H_ * sizeof(float);     // 128 MiB
  __bf16* Xpack = (__bf16*)(ws + off); off += M1 * KIN * 2;               //  32 MiB
  __bf16* Upack = (__bf16*)(ws + off); off += (size_t)KIN * H_ * 2;       //   1 MiB
  __bf16* Wpack = (__bf16*)(ws + off); off += (size_t)H_ * H_ * 2;        //   2 MiB
  __bf16* Vpack = (__bf16*)(ws + off);                                    //   1 MiB

  pack_a_bf16<<<(int)((M1 * KIN + 255) / 256), 256, 0, stream>>>(X, Xpack, (int)M1, KIN);
  pack_b_bf16<<<(KIN * H_ + 255) / 256, 256, 0, stream>>>(U, Upack, KIN, H_);
  pack_b_bf16<<<(H_ * H_ + 255) / 256, 256, 0, stream>>>(W, Wpack, H_, H_);
  pack_b_bf16<<<(H_ * O_ + 255) / 256, 256, 0, stream>>>(V, Vpack, H_, O_);

  dim3 g1(B_ * T_ / 128, H_ / 128);
  xu_gemm<<<g1, 256, 0, stream>>>(Xpack, Upack, bb, XU);

  rnn_scan<<<B_ / 16, 1024, 0, stream>>>(h0, Wpack, Vpack, XU, cc, Y);
}
